// Edge_78408922956191
// MI455X (gfx1250) — matrix operand order repaired and checked by reading, not compiled, not run on hardware
//
#include <hip/hip_runtime.h>
#include <hip/hip_bf16.h>

// ---------------------------------------------------------------------------
// DGCNN-style network for MI455X (gfx1250, wave32, WMMA f32_16x16x32_f16).
// All GEMMs run through v_wmma_f32_16x16x32_f16 with fp16 activations and
// BN affines folded into pre-packed (fragment-layout) fp16 weights.
// GEMM wave tile: 32 rows x 64 cols (2 A frags x 4 B frags = 8 WMMAs/ktile),
// so each B fragment load feeds two WMMAs.
// ---------------------------------------------------------------------------

typedef _Float16 h16;
typedef __attribute__((ext_vector_type(16))) _Float16 v16h;
typedef __attribute__((ext_vector_type(8)))  _Float16 v8h;
typedef __attribute__((ext_vector_type(8)))  float    v8f;

constexpr int kB   = 128;            // graphs
constexpr int kN   = 128;            // nodes per graph
constexpr int kK   = 20;             // neighbors
constexpr int kGN  = kB * kN;        // 16384 node rows
constexpr long kM1 = (long)kGN * kK; // 327680 edge rows
constexpr int kMIX = 10;
constexpr float kEPS = 1e-5f;

union AFrag { v16h v; v8h h2[2]; };

// ---------------- pairwise distances -----------------------------------

__global__ __launch_bounds__(256) void dist_pos_kernel(
    const float* __restrict__ pos, float* __restrict__ D) {
  int id = blockIdx.x * 256 + threadIdx.x;        // b*N*N + i*N + j
  int j  = id & (kN - 1);
  int bi = id >> 7;                                // b*N + i
  int b  = bi >> 7;
  const float* pi = pos + (size_t)bi * 3;
  const float* pj = pos + ((size_t)(b * kN + j)) * 3;
  float dx = pi[0] - pj[0], dy = pi[1] - pj[1], dz = pi[2] - pj[2];
  D[id] = dx * dx + dy * dy + dz * dz;
}

__global__ __launch_bounds__(256) void dist_f16_kernel(
    const h16* __restrict__ X, float* __restrict__ D) {
  int id = blockIdx.x * 256 + threadIdx.x;
  int j  = id & (kN - 1);
  int bi = id >> 7;
  int b  = bi >> 7;
  const h16* xi = X + (size_t)bi * 192;            // concat buffer, ld=192
  const h16* xj = X + ((size_t)(b * kN + j)) * 192;
  float acc = 0.f;
  for (int c = 0; c < 64; ++c) {
    float d = (float)xi[c] - (float)xj[c];
    acc += d * d;
  }
  D[id] = acc;
}

// ---------------- 20-NN selection (lexicographic, matches top_k ties) ---

__global__ __launch_bounds__(256) void knn_sel_kernel(
    const float* __restrict__ D, int* __restrict__ idx) {
  int id = blockIdx.x * 256 + threadIdx.x;         // b*N + i
  if (id >= kGN) return;
  const float* row = D + (size_t)id * kN;
  int* out = idx + (size_t)id * kK;
  float lastd = -1e38f; int lastj = -1;
  for (int k = 0; k < kK; ++k) {
    float bd = 1e38f; int bj = 0;
    for (int j = 0; j < kN; ++j) {
      float d = row[j];
      bool cand = (d > lastd) || (d == lastd && j > lastj);
      if (cand && d < bd) { bd = d; bj = j; }
    }
    out[k] = bj; lastd = bd; lastj = bj;
  }
}

// ---------------- first edge layer: [xi, xj-xi](6) -> 64, VALU ----------

__global__ __launch_bounds__(256) void edge0_kernel(
    const float* __restrict__ pos, const int* __restrict__ idx,
    const float* __restrict__ W0, const float* __restrict__ b0,
    h16* __restrict__ h0) {
  __shared__ float ws[6 * 64];
  __shared__ float bs[64];
  for (int i = threadIdx.x; i < 384; i += 256) ws[i] = W0[i];
  if (threadIdx.x < 64) bs[threadIdx.x] = b0[threadIdx.x];
  __syncthreads();
  long r = (long)blockIdx.x * 256 + threadIdx.x;   // edge row < kM1 (exact)
  int g = (int)(r / kK);                            // b*N + i
  int b = g >> 7;
  int j = idx[r];
  const float* pi = pos + (size_t)g * 3;
  const float* pj = pos + ((size_t)(b * kN + j)) * 3;
  float e[6];
  e[0] = pi[0]; e[1] = pi[1]; e[2] = pi[2];
  e[3] = pj[0] - pi[0]; e[4] = pj[1] - pi[1]; e[5] = pj[2] - pi[2];
  h16* o = h0 + r * 64;
  for (int c = 0; c < 64; ++c) {
    float acc = bs[c];
#pragma unroll
    for (int f = 0; f < 6; ++f) acc += e[f] * ws[f * 64 + c];
    o[c] = (h16)fmaxf(acc, 0.f);
  }
}

// ---------------- BN statistics: one block per channel ------------------

__global__ __launch_bounds__(256) void stats_kernel(
    const h16* __restrict__ h, long M, int C,
    const float* __restrict__ gamma, const float* __restrict__ beta,
    float* __restrict__ s, float* __restrict__ t) {
  int c = blockIdx.x;
  __shared__ float ss[256], sq[256];
  float a = 0.f, q = 0.f;
  for (long r = threadIdx.x; r < M; r += 256) {
    float v = (float)h[r * C + c];
    a += v; q += v * v;
  }
  ss[threadIdx.x] = a; sq[threadIdx.x] = q;
  __syncthreads();
  for (int st = 128; st > 0; st >>= 1) {
    if ((int)threadIdx.x < st) {
      ss[threadIdx.x] += ss[threadIdx.x + st];
      sq[threadIdx.x] += sq[threadIdx.x + st];
    }
    __syncthreads();
  }
  if (threadIdx.x == 0) {
    float inv = 1.f / (float)M;
    float mean = ss[0] * inv;
    float var  = fmaxf(sq[0] * inv - mean * mean, 0.f);
    float sc   = gamma[c] * rsqrtf(var + kEPS);
    s[c] = sc;
    t[c] = beta[c] - mean * sc;
  }
}

// ---------------- pack f32 weights into WMMA B-fragment order -----------
// B fragment (16-bit, 32x16): lane l -> n = l&15, K = (l>>4)*16 + e, e=0..15.
// Packed index: ((kt*NT + nt)*32 + lane)*16 + e  -> 32B contiguous per lane.

__global__ __launch_bounds__(256) void pack_w_kernel(
    const float* __restrict__ W, const float* __restrict__ s,
    int Kdim, int Ndim, h16* __restrict__ Wp) {
  int idx = blockIdx.x * 256 + threadIdx.x;
  if (idx >= Kdim * Ndim) return;
  int e = idx & 15, l = (idx >> 4) & 31, tile = idx >> 9;
  int NT = Ndim >> 4;
  int nt = tile % NT, kt = tile / NT;
  int k = kt * 32 + (l >> 4) * 16 + e;
  int n = nt * 16 + (l & 15);
  float sv = s ? s[k] : 1.f;
  Wp[idx] = (h16)(sv * W[(size_t)k * Ndim + n]);
}

__global__ __launch_bounds__(256) void pack_bias_kernel(
    const float* __restrict__ W, const float* __restrict__ b,
    const float* __restrict__ t, int Kdim, int Ndim, float* __restrict__ bp) {
  int n = blockIdx.x * 256 + threadIdx.x;
  if (n >= Ndim) return;
  float acc = b[n];
  if (t)
    for (int k = 0; k < Kdim; ++k) acc += t[k] * W[(size_t)k * Ndim + n];
  bp[n] = acc;
}

// ---------------- WMMA GEMM: Out = relu(A @ Wp + bp), fp16 in/out -------
// 8 waves/block; wave handles 32 rows x 64 cols: 2 A frags x 4 B frags,
// 8 WMMAs per k-tile, each B fragment reused twice.
// grid = (M/256, Ndim/64).

__global__ __launch_bounds__(256) void wmma_gemm_relu(
    const h16* __restrict__ A, const h16* __restrict__ Wp,
    const float* __restrict__ bp, h16* __restrict__ Out,
    int Kdim, int Ndim) {
  const int lane = threadIdx.x & 31;
  const int wave = threadIdx.x >> 5;
  const long rowBase = (long)blockIdx.x * 256 + wave * 32;
  const int NT = Ndim >> 4;
  const int ntBase = blockIdx.y * 4;
  const int g = lane >> 4;          // A-fragment K-group
  const int ml = lane & 15;         // A row within tile
  const h16* aptr0 = A + (rowBase + ml) * (long)Kdim;
  const h16* aptr1 = aptr0 + 16 * (long)Kdim;
  v8f a00 = {}, a01 = {}, a02 = {}, a03 = {};
  v8f a10 = {}, a11 = {}, a12 = {}, a13 = {};
  const int KT = Kdim >> 5;
  for (int kt = 0; kt < KT; ++kt) {
    AFrag af0, af1;
    // lanes 0-15: K = [0..15]; lanes 16-31: K = [16..31]
    af0.h2[0] = *(const v8h*)(aptr0 + kt * 32 + g * 16);
    af0.h2[1] = *(const v8h*)(aptr0 + kt * 32 + 8 + g * 16);
    af1.h2[0] = *(const v8h*)(aptr1 + kt * 32 + g * 16);
    af1.h2[1] = *(const v8h*)(aptr1 + kt * 32 + 8 + g * 16);
    if (kt + 1 < KT) {               // prefetch next A k-tile
      __builtin_prefetch(aptr0 + (kt + 1) * 32, 0, 1);
      __builtin_prefetch(aptr1 + (kt + 1) * 32, 0, 1);
    }
    const h16* bbase = Wp + ((long)kt * NT + ntBase) * 512 + lane * 16;
    v16h bf0 = *(const v16h*)(bbase);
    v16h bf1 = *(const v16h*)(bbase + 512);
    v16h bf2 = *(const v16h*)(bbase + 1024);
    v16h bf3 = *(const v16h*)(bbase + 1536);
    a00 = __builtin_amdgcn_wmma_f32_16x16x32_f16(false, af0.v, false, bf0,
                                                 (short)0, a00, false, false);
    a10 = __builtin_amdgcn_wmma_f32_16x16x32_f16(false, af1.v, false, bf0,
                                                 (short)0, a10, false, false);
    a01 = __builtin_amdgcn_wmma_f32_16x16x32_f16(false, af0.v, false, bf1,
                                                 (short)0, a01, false, false);
    a11 = __builtin_amdgcn_wmma_f32_16x16x32_f16(false, af1.v, false, bf1,
                                                 (short)0, a11, false, false);
    a02 = __builtin_amdgcn_wmma_f32_16x16x32_f16(false, af0.v, false, bf2,
                                                 (short)0, a02, false, false);
    a12 = __builtin_amdgcn_wmma_f32_16x16x32_f16(false, af1.v, false, bf2,
                                                 (short)0, a12, false, false);
    a03 = __builtin_amdgcn_wmma_f32_16x16x32_f16(false, af0.v, false, bf3,
                                                 (short)0, a03, false, false);
    a13 = __builtin_amdgcn_wmma_f32_16x16x32_f16(false, af1.v, false, bf3,
                                                 (short)0, a13, false, false);
  }
  // C/D layout: VGPR r -> rows (lane>>4)*8 + r, col = nt*16 + (lane&15)
  const int nl = lane & 15;
  const int rg = (lane >> 4) * 8;
#pragma unroll
  for (int jj = 0; jj < 4; ++jj) {
    v8f r0 = (jj == 0) ? a00 : (jj == 1) ? a01 : (jj == 2) ? a02 : a03;
    v8f r1 = (jj == 0) ? a10 : (jj == 1) ? a11 : (jj == 2) ? a12 : a13;
    int col = (ntBase + jj) * 16 + nl;
    float bb = bp[col];
    h16* o0 = Out + (rowBase + rg) * (long)Ndim + col;
    h16* o1 = o0 + 16 * (long)Ndim;
#pragma unroll
    for (int r = 0; r < 8; ++r) {
      o0[(long)r * Ndim] = (h16)fmaxf(r0[r] + bb, 0.f);
      o1[(long)r * Ndim] = (h16)fmaxf(r1[r] + bb, 0.f);
    }
  }
}

// ---------------- max aggregation over K with BN affine -----------------

__global__ __launch_bounds__(256) void maxagg_kernel(
    const h16* __restrict__ h, int C, const float* __restrict__ s,
    const float* __restrict__ t, h16* __restrict__ out, int ldOut, int colOff) {
  long id = (long)blockIdx.x * 256 + threadIdx.x;  // < kGN*C (exact)
  int c = (int)(id % C);
  long g = id / C;
  const h16* p = h + g * kK * C + c;
  float sc = s[c], sh = t[c];
  float best = -1e38f;
  for (int k = 0; k < kK; ++k)
    best = fmaxf(best, sc * (float)p[(long)k * C] + sh);
  out[g * ldOut + colOff + c] = (h16)best;
}

// ---------------- build conv2 edge features [xi, xj-xi] -----------------

__global__ __launch_bounds__(256) void e2build_kernel(
    const h16* __restrict__ X, const int* __restrict__ idx,
    h16* __restrict__ e2) {
  long id = (long)blockIdx.x * 256 + threadIdx.x;  // < kM1*64 (exact)
  int c = (int)(id & 63);
  long r = id >> 6;
  int g = (int)(r / kK);
  int b = g >> 7;
  int j = idx[r];
  float xi = (float)X[(size_t)g * 192 + c];
  float xj = (float)X[((size_t)(b * kN + j)) * 192 + c];
  e2[r * 128 + c] = (h16)xi;
  e2[r * 128 + 64 + c] = (h16)(xj - xi);
}

// ---------------- mixture tail: 256 -> 1 dot, running mean/var ----------

__global__ __launch_bounds__(256) void mixfinal_kernel(
    const h16* __restrict__ m1, const float* __restrict__ w2,
    const float* __restrict__ b2, float* __restrict__ sum,
    float* __restrict__ sumsq, int first) {
  int id = blockIdx.x * 256 + threadIdx.x;
  if (id >= kGN) return;
  float acc = b2[0];
  const h16* row = m1 + (size_t)id * 256;
  for (int c = 0; c < 256; ++c) acc += (float)row[c] * w2[c];
  if (first) { sum[id] = acc; sumsq[id] = acc * acc; }
  else       { sum[id] += acc; sumsq[id] += acc * acc; }
}

__global__ __launch_bounds__(256) void finalize_kernel(
    const float* __restrict__ sum, const float* __restrict__ sumsq,
    float* __restrict__ out) {
  int id = blockIdx.x * 256 + threadIdx.x;
  if (id >= kGN) return;
  float inv = 1.f / (float)kMIX;
  float mu = sum[id] * inv;
  float var = fmaxf(sumsq[id] * inv - mu * mu, 0.f);
  out[id] = mu;
  out[kGN + id] = sqrtf(var);
}

// ---------------------------------------------------------------------------

extern "C" void kernel_launch(void* const* d_in, const int* in_sizes, int n_in,
                              void* d_out, int out_size, void* d_ws,
                              size_t ws_size, hipStream_t stream) {
  (void)in_sizes; (void)n_in; (void)out_size; (void)ws_size;

  const float* pos    = (const float*)d_in[0];
  const float* c1_w0  = (const float*)d_in[1];
  const float* c1_b0  = (const float*)d_in[2];
  const float* c1_0g  = (const float*)d_in[3];
  const float* c1_0b  = (const float*)d_in[4];
  const float* c1_w1  = (const float*)d_in[5];
  const float* c1_b1  = (const float*)d_in[6];
  const float* c1_1g  = (const float*)d_in[7];
  const float* c1_1b  = (const float*)d_in[8];
  const float* c1_w2  = (const float*)d_in[9];
  const float* c1_b2  = (const float*)d_in[10];
  const float* c1_2g  = (const float*)d_in[11];
  const float* c1_2b  = (const float*)d_in[12];
  const float* c2_w0  = (const float*)d_in[13];
  const float* c2_b0  = (const float*)d_in[14];
  const float* c2_0g  = (const float*)d_in[15];
  const float* c2_0b  = (const float*)d_in[16];
  const float* l1_w   = (const float*)d_in[17];
  const float* l1_b   = (const float*)d_in[18];
  const float* l1_g   = (const float*)d_in[19];
  const float* l1_be  = (const float*)d_in[20];
  const float* mix_w0 = (const float*)d_in[21];
  const float* mix_b0 = (const float*)d_in[22];
  const float* mix_w1 = (const float*)d_in[23];
  const float* mix_b1 = (const float*)d_in[24];
  const float* mix_w2 = (const float*)d_in[25];
  const float* mix_b2 = (const float*)d_in[26];
  // d_in[27] = input_idx, unused in eval path.
  float* out = (float*)d_out;

  // ---- workspace bump allocator (aligned 256B) ----
  uint8_t* wsb = (uint8_t*)d_ws;
  size_t off = 0;
  auto alloc = [&](size_t bytes) -> void* {
    off = (off + 255) & ~(size_t)255;
    void* p = wsb + off;
    off += bytes;
    return p;
  };
  int*   idx1 = (int*)alloc((size_t)kM1 * 4);
  int*   idx2 = (int*)alloc((size_t)kM1 * 4);
  float* D    = (float*)alloc((size_t)kB * kN * kN * 4);
  h16*   hcat = (h16*)alloc((size_t)kGN * 192 * 2);   // [x1 | x2] concat
  h16*   hl1  = (h16*)alloc((size_t)kGN * 256 * 2);
  h16*   m0   = (h16*)alloc((size_t)kGN * 256 * 2);
  h16*   m1b  = (h16*)alloc((size_t)kGN * 256 * 2);
  float* sumH = (float*)alloc((size_t)kGN * 4);
  float* sqH  = (float*)alloc((size_t)kGN * 4);
  float* s0 = (float*)alloc(64 * 4);  float* t0 = (float*)alloc(64 * 4);
  float* s1 = (float*)alloc(64 * 4);  float* t1 = (float*)alloc(64 * 4);
  float* s2 = (float*)alloc(64 * 4);  float* t2 = (float*)alloc(64 * 4);
  float* s3 = (float*)alloc(128 * 4); float* t3 = (float*)alloc(128 * 4);
  float* sl = (float*)alloc(256 * 4); float* tl = (float*)alloc(256 * 4);
  float* bp1  = (float*)alloc(64 * 4);
  float* bp2  = (float*)alloc(64 * 4);
  float* bp3  = (float*)alloc(128 * 4);
  float* bpl1 = (float*)alloc(256 * 4);
  float* bpm0 = (float*)alloc((size_t)kMIX * 256 * 4);
  float* bpm1 = (float*)alloc((size_t)kMIX * 256 * 4);
  h16* Wp1  = (h16*)alloc((size_t)64 * 64 * 2);
  h16* Wp2  = (h16*)alloc((size_t)64 * 64 * 2);
  h16* Wp3  = (h16*)alloc((size_t)128 * 128 * 2);
  h16* Wpl1 = (h16*)alloc((size_t)192 * 256 * 2);
  h16* Wpm0 = (h16*)alloc((size_t)kMIX * 256 * 256 * 2);
  h16* Wpm1 = (h16*)alloc((size_t)kMIX * 256 * 256 * 2);
  // Region R0: h0|h1, later reused as e2 (sizes match exactly).
  h16* h0 = (h16*)alloc((size_t)kM1 * 64 * 2 * 2);
  h16* h1 = h0 + (size_t)kM1 * 64;
  h16* e2 = h0;
  // Region R1: h2 (first half), later reused as h3 (full).
  h16* h2 = (h16*)alloc((size_t)kM1 * 128 * 2);
  h16* h3 = h2;

  const int distBlocks = (kB * kN * kN) / 256;       // 8192
  const int selBlocks  = kGN / 256;                  // 64
  const int gemmM1     = (int)(kM1 / 256);           // 1280
  const int gemmMN     = kGN / 256;                  // 64

  // ---- EdgeConv1 ----
  dist_pos_kernel<<<distBlocks, 256, 0, stream>>>(pos, D);
  knn_sel_kernel<<<selBlocks, 256, 0, stream>>>(D, idx1);
  edge0_kernel<<<(int)(kM1 / 256), 256, 0, stream>>>(pos, idx1, c1_w0, c1_b0, h0);

  stats_kernel<<<64, 256, 0, stream>>>(h0, kM1, 64, c1_0g, c1_0b, s0, t0);
  pack_w_kernel<<<(64 * 64) / 256, 256, 0, stream>>>(c1_w1, s0, 64, 64, Wp1);
  pack_bias_kernel<<<1, 256, 0, stream>>>(c1_w1, c1_b1, t0, 64, 64, bp1);
  wmma_gemm_relu<<<dim3(gemmM1, 1), 256, 0, stream>>>(h0, Wp1, bp1, h1, 64, 64);

  stats_kernel<<<64, 256, 0, stream>>>(h1, kM1, 64, c1_1g, c1_1b, s1, t1);
  pack_w_kernel<<<(64 * 64) / 256, 256, 0, stream>>>(c1_w2, s1, 64, 64, Wp2);
  pack_bias_kernel<<<1, 256, 0, stream>>>(c1_w2, c1_b2, t1, 64, 64, bp2);
  wmma_gemm_relu<<<dim3(gemmM1, 1), 256, 0, stream>>>(h1, Wp2, bp2, h2, 64, 64);

  stats_kernel<<<64, 256, 0, stream>>>(h2, kM1, 64, c1_2g, c1_2b, s2, t2);
  maxagg_kernel<<<(int)((long)kGN * 64 / 256), 256, 0, stream>>>(h2, 64, s2, t2, hcat, 192, 0);

  // ---- EdgeConv2 ----
  dist_f16_kernel<<<distBlocks, 256, 0, stream>>>(hcat, D);
  knn_sel_kernel<<<selBlocks, 256, 0, stream>>>(D, idx2);
  e2build_kernel<<<(int)(kM1 * 64 / 256), 256, 0, stream>>>(hcat, idx2, e2);

  pack_w_kernel<<<(128 * 128) / 256, 256, 0, stream>>>(c2_w0, nullptr, 128, 128, Wp3);
  pack_bias_kernel<<<1, 256, 0, stream>>>(c2_w0, c2_b0, nullptr, 128, 128, bp3);
  wmma_gemm_relu<<<dim3(gemmM1, 2), 256, 0, stream>>>(e2, Wp3, bp3, h3, 128, 128);

  stats_kernel<<<128, 256, 0, stream>>>(h3, kM1, 128, c2_0g, c2_0b, s3, t3);
  maxagg_kernel<<<(int)((long)kGN * 128 / 256), 256, 0, stream>>>(h3, 128, s3, t3, hcat, 192, 64);

  // ---- fuse layer (192 -> 256) ----
  pack_w_kernel<<<(192 * 256 + 255) / 256, 256, 0, stream>>>(l1_w, nullptr, 192, 256, Wpl1);
  pack_bias_kernel<<<1, 256, 0, stream>>>(l1_w, l1_b, nullptr, 192, 256, bpl1);
  wmma_gemm_relu<<<dim3(gemmMN, 4), 256, 0, stream>>>(hcat, Wpl1, bpl1, hl1, 192, 256);
  stats_kernel<<<256, 256, 0, stream>>>(hl1, kGN, 256, l1_g, l1_be, sl, tl);

  // ---- mixture heads ----
  for (int hh = 0; hh < kMIX; ++hh) {
    const float* w0h = mix_w0 + (size_t)hh * 256 * 256;
    const float* w1h = mix_w1 + (size_t)hh * 256 * 256;
    h16* Wp0h = Wpm0 + (size_t)hh * 256 * 256;
    h16* Wp1h = Wpm1 + (size_t)hh * 256 * 256;
    float* bp0h = bpm0 + (size_t)hh * 256;
    float* bp1h = bpm1 + (size_t)hh * 256;
    pack_w_kernel<<<(256 * 256) / 256, 256, 0, stream>>>(w0h, sl, 256, 256, Wp0h);
    pack_bias_kernel<<<1, 256, 0, stream>>>(w0h, mix_b0 + (size_t)hh * 256, tl, 256, 256, bp0h);
    pack_w_kernel<<<(256 * 256) / 256, 256, 0, stream>>>(w1h, nullptr, 256, 256, Wp1h);
    pack_bias_kernel<<<1, 256, 0, stream>>>(w1h, mix_b1 + (size_t)hh * 256, nullptr, 256, 256, bp1h);
    wmma_gemm_relu<<<dim3(gemmMN, 4), 256, 0, stream>>>(hl1, Wp0h, bp0h, m0, 256, 256);
    wmma_gemm_relu<<<dim3(gemmMN, 4), 256, 0, stream>>>(m0, Wp1h, bp1h, m1b, 256, 256);
    mixfinal_kernel<<<selBlocks, 256, 0, stream>>>(m1b, mix_w2 + (size_t)hh * 256,
                                                   mix_b2 + hh, sumH, sqH, hh == 0);
  }
  finalize_kernel<<<selBlocks, 256, 0, stream>>>(sumH, sqH, out);
}
